// PCAMoENet_26766236189238
// MI455X (gfx1250) — matrix operand order repaired and checked by reading, not compiled, not run on hardware
//
#include <hip/hip_runtime.h>
#include <hip/hip_bf16.h>

// ============================================================================
// PCAMoENet forward for MI455X (gfx1250), wave32 + WMMA f32_16x16x32_f16.
// Expert convs = per-patch GEMMs: weights aggregated (Sum_k v*W_k) to f16
// scratch, then WMMA over the 1024 pixels of each 32x32 patch.
// LDS tiles are stored K-contiguous per lane so fragment loads lower to
// ds_load_b128 (2 per fragment) instead of 16x ds_load_u16 + v_mov_b16 packs.
// ============================================================================

typedef __attribute__((ext_vector_type(16))) _Float16 v16h;
typedef __attribute__((ext_vector_type(8)))  _Float16 v8h;
typedef __attribute__((ext_vector_type(8)))  float    v8f;

#define PATCH 32
#define LDW3 296   // 3x3 weight tile row stride (halves): 9*32=288 (+8 pad)
#define LDX3 40    // 3x3 act tile per-pixel channel stride: 32 (+8 pad)

// ---------------- WMMA fragment helpers (contiguous k order: k = 16*(lane/16) + s) ----------
// All loads are 16B-aligned and K-contiguous -> each v8h = one ds_load_b128.

__device__ inline v16h cat16(v8h lo, v8h hi) {
  return __builtin_shufflevector(lo, hi, 0, 1, 2, 3, 4, 5, 6, 7,
                                 8, 9, 10, 11, 12, 13, 14, 15);
}

__device__ inline v16h frag_a(const _Float16* A, int rs, int m0, int k0) {
  // A 16x32 f16: lane = row; elems = K kb..kb+15 (contiguous);
  // kb gets +16 for lanes 16-31. rs must be a multiple of 8 halves.
  int lane = threadIdx.x & 31;
  int m = m0 + (lane & 15);
  int kb = k0 + ((lane >> 4) << 4);
  const _Float16* p = A + m * rs + kb;
  return cat16(*(const v8h*)p, *(const v8h*)(p + 8));
}

__device__ inline v16h frag_b(const _Float16* Bm, int ldk, int k0, int n0) {
  // B 32x16 f16: lane = column n; elems = K rows kb..kb+15 (lanes>=16 -> +16).
  // Bm is pixel-major [n][k]; ldk multiple of 8 halves.
  int lane = threadIdx.x & 31;
  int n = n0 + (lane & 15);
  int kb = k0 + ((lane >> 4) << 4);
  const _Float16* p = Bm + n * ldk + kb;
  return cat16(*(const v8h*)p, *(const v8h*)(p + 8));
}

__device__ inline v16h frag_bx(const _Float16* X, int n0, int dy, int dx) {
  // X: [(6 rows)*(36 cols)][40 ch] zero-padded patch slab, pixel n in [0,128)
  int lane = threadIdx.x & 31;
  int n = n0 + (lane & 15);
  int row = (n >> 5) + dy;
  int col = (n & 31) + dx;
  int kb = (lane >> 4) << 4;
  const _Float16* p = X + (row * 36 + col) * LDX3 + kb;
  return cat16(*(const v8h*)p, *(const v8h*)(p + 8));
}

__device__ inline v8f wmma16(v16h a, v16h b, v8f c) {
  return __builtin_amdgcn_wmma_f32_16x16x32_f16(false, a, false, b, (short)0, c,
                                                false, false);
}

// ---------------- expert weight aggregation (per patch, -> f16) -------------
// Output layout is TAP-MAJOR: Wout[np][o][tap][i] so the GEMM A-fragments read
// contiguous channels for a fixed tap (khw=1 for 1x1 convs).

__global__ void k_agg_w(const float* __restrict__ v, const float* __restrict__ Wexp,
                        const float* __restrict__ bexp, _Float16* __restrict__ Wout,
                        float* __restrict__ bout, int O, int I, int khw,
                        int hp, int wp) {
  int np = blockIdx.y;
  int ppb = hp * wp;
  int b = np / ppb, pr = (np % ppb) / wp, pc = (np % ppb) % wp;
  float vk[5];
#pragma unroll
  for (int k = 0; k < 5; ++k) vk[k] = v[((b * 5 + k) * hp + pr) * wp + pc];
  long tot = (long)O * I * khw;
  for (long idx = (long)blockIdx.x * 256 + threadIdx.x; idx < tot;
       idx += (long)gridDim.x * 256) {
    int i = (int)(idx % I);
    long r = idx / I;
    int t = (int)(r % khw);
    int o = (int)(r / khw);
    long src = ((long)o * I + i) * khw + t;   // Wexp: [k][O][I][kh*kw]
    float s = 0.f;
#pragma unroll
    for (int k = 0; k < 5; ++k) s += vk[k] * Wexp[(long)k * tot + src];
    Wout[(long)np * tot + idx] = (_Float16)s;
  }
  for (long o = (long)blockIdx.x * 256 + threadIdx.x; o < O;
       o += (long)gridDim.x * 256) {
    float s = 0.f;
#pragma unroll
    for (int k = 0; k < 5; ++k) s += vk[k] * bexp[k * O + o];
    bout[(long)np * O + o] = s;
  }
}

// ---------------- 1x1 conv as per-patch WMMA GEMM ---------------------------
// grid = (numPatches, O/64); block = 256 (8 waves). out[O,1024] = W[O,I]*x[I,1024]

__global__ void __launch_bounds__(256)
k_conv1x1_wmma(const float* __restrict__ x, const _Float16* __restrict__ Wg,
               const float* __restrict__ bg, float* __restrict__ y,
               int I, int O, int H, int W, int sharedW) {
  __shared__ __align__(16) _Float16 sA[64 * 136];    // weights  [o][i], ld=I+8
  __shared__ __align__(16) _Float16 sB[128 * 136];   // act (pixel-major) [p][i]
  int np = blockIdx.x, ot = blockIdx.y;
  int hp = H / PATCH, wp = W / PATCH, ppb = hp * wp;
  int b = np / ppb, pr = (np % ppb) / wp, pc = (np % ppb) % wp;
  int h0 = pr * PATCH, w0 = pc * PATCH;
  int oBase = ot * 64;
  int tid = threadIdx.x, wave = tid >> 5;
  int ldk = I + 8;                                   // multiple of 8 halves
  long HW = (long)H * W;
  const _Float16* Wp = Wg + (sharedW ? 0L : (long)np * O * I);

  for (int idx = tid; idx < 64 * I; idx += 256) {
    int o = idx / I, i = idx - o * I;
    int og = oBase + o;
    sA[o * ldk + i] = (og < O) ? Wp[(long)og * I + i] : (_Float16)0.f;
  }
  int mi = wave & 3, nh = wave >> 2;
  int lane = tid & 31, nLane = lane & 15, mOff = (lane >> 4) << 3;

  for (int nc = 0; nc < 8; ++nc) {          // 8 chunks of 128 pixels
    __syncthreads();
    for (int idx = tid; idx < I * 128; idx += 256) {
      int i = idx >> 7, q = idx & 127;
      int p = nc * 128 + q;
      int h = h0 + (p >> 5), w = w0 + (p & 31);
      sB[q * ldk + i] = (_Float16)x[((long)(b * I + i)) * HW + (long)h * W + w];
    }
    if (nc + 1 < 8) {                       // gfx1250 global_prefetch_b8
      int q = tid & 127, i = tid >> 7;
      int p = (nc + 1) * 128 + q;
      int h = h0 + (p >> 5), w = w0 + (p & 31);
      if (i < I)
        __builtin_prefetch(&x[((long)(b * I + i)) * HW + (long)h * W + w], 0, 1);
    }
    __syncthreads();
    v8f acc[4] = {};
    for (int k0 = 0; k0 < I; k0 += 32) {
      v16h a = frag_a(sA, ldk, mi * 16, k0);
#pragma unroll
      for (int j = 0; j < 4; ++j) {
        v16h bm = frag_b(sB, ldk, k0, nh * 64 + j * 16);
        acc[j] = wmma16(a, bm, acc[j]);
      }
    }
#pragma unroll
    for (int j = 0; j < 4; ++j) {
      int p = nc * 128 + nh * 64 + j * 16 + nLane;
      int h = h0 + (p >> 5), w = w0 + (p & 31);
#pragma unroll
      for (int e = 0; e < 8; ++e) {
        int og = oBase + mi * 16 + mOff + e;
        if (og < O) {
          float bias = bg ? bg[(sharedW ? 0L : (long)np * O) + og] : 0.f;
          y[((long)(b * O + og)) * HW + (long)h * W + w] = acc[j][e] + bias;
        }
      }
    }
  }
}

// ---------------- 3x3 expert conv (pad=1 inside patch) as 9 shifted WMMAs ---

__global__ void __launch_bounds__(256)
k_conv3x3_wmma(const float* __restrict__ x, const _Float16* __restrict__ Wg,
               const float* __restrict__ bg, float* __restrict__ y,
               int I, int O, int H, int W) {
  __shared__ __align__(16) _Float16 sW[64 * LDW3];      // [o][tap][32ch]
  __shared__ __align__(16) _Float16 sX[6 * 36 * LDX3];  // [(row,col)][32ch]
  int np = blockIdx.x, ot = blockIdx.y;
  int hp = H / PATCH, wp = W / PATCH, ppb = hp * wp;
  int b = np / ppb, pr = (np % ppb) / wp, pc = (np % ppb) % wp;
  int h0 = pr * PATCH, w0 = pc * PATCH;
  int oBase = ot * 64;
  int tid = threadIdx.x, wave = tid >> 5;
  long HW = (long)H * W;
  long IK = (long)I * 9;
  const _Float16* Wp = Wg + (long)np * O * IK;   // tap-major: [o][9][I]
  int mi = wave & 3, nh = wave >> 2;
  int lane = tid & 31, nLane = lane & 15, mOff = (lane >> 4) << 3;

  for (int nc = 0; nc < 8; ++nc) {          // 4 patch rows per chunk
    int r0 = nc * 4;
    v8f acc[4] = {};
    for (int ic = 0; ic < I; ic += 32) {    // K (input-channel) chunks
      __syncthreads();
      for (int idx = tid; idx < 64 * 288; idx += 256) {
        int o = idx / 288, rr = idx - o * 288;
        int t = rr >> 5, il = rr & 31;
        int og = oBase + o;
        sW[o * LDW3 + rr] =
            (og < O) ? Wp[((long)og * 9 + t) * I + ic + il] : (_Float16)0.f;
      }
      for (int idx = tid; idx < 32 * 6 * 36; idx += 256) {
        int i = idx / 216, rem = idx - i * 216;
        int rl = rem / 36, cl = rem - rl * 36;
        int prl = r0 + rl - 1, pcl = cl - 1;
        _Float16 vv = (_Float16)0.f;
        if (prl >= 0 && prl < PATCH && pcl >= 0 && pcl < PATCH)
          vv = (_Float16)x[((long)(b * I + ic + i)) * HW +
                           (long)(h0 + prl) * W + (w0 + pcl)];
        sX[(rl * 36 + cl) * LDX3 + i] = vv;
      }
      __syncthreads();
#pragma unroll
      for (int dy = 0; dy < 3; ++dy)
#pragma unroll
        for (int dx = 0; dx < 3; ++dx) {
          v16h a = frag_a(sW + (dy * 3 + dx) * 32, LDW3, mi * 16, 0);
#pragma unroll
          for (int j = 0; j < 4; ++j) {
            v16h bm = frag_bx(sX, nh * 64 + j * 16, dy, dx);
            acc[j] = wmma16(a, bm, acc[j]);
          }
        }
    }
#pragma unroll
    for (int j = 0; j < 4; ++j) {
      int p = nc * 128 + nh * 64 + j * 16 + nLane;
      int h = h0 + (p >> 5), w = w0 + (p & 31);
#pragma unroll
      for (int e = 0; e < 8; ++e) {
        int og = oBase + mi * 16 + mOff + e;
        if (og < O)
          y[((long)(b * O + og)) * HW + (long)h * W + w] =
              acc[j][e] + bg[(long)np * O + og];
      }
    }
  }
}

// ---------------- streaming / small kernels ---------------------------------

__global__ void k_layernorm(const float* __restrict__ x, const float* __restrict__ w,
                            const float* __restrict__ bb, float* __restrict__ y,
                            int C, long HW) {
  long p = (long)blockIdx.x * 256 + threadIdx.x;
  if (p >= HW) return;
  int b = blockIdx.y;
  const float* xb = x + (long)b * C * HW;
  float mu = 0.f, m2 = 0.f;
  for (int c = 0; c < C; ++c) { float t = xb[c * HW + p]; mu += t; m2 += t * t; }
  mu /= C;
  float inv = rsqrtf(m2 / C - mu * mu + 1e-6f);
  float* yb = y + (long)b * C * HW;
  for (int c = 0; c < C; ++c)
    yb[c * HW + p] = (xb[c * HW + p] - mu) * inv * w[c] + bb[c];
}

__global__ void k_gate(const float* __restrict__ x, float* __restrict__ y,
                       int C, long HW, long total) {
  long idx = (long)blockIdx.x * 256 + threadIdx.x;
  if (idx >= total) return;
  long b = idx / (C * HW);
  long r = idx - b * C * HW;
  long c = r / HW, p = r - c * HW;
  const float* xb = x + (b * 2 * C + c) * HW + p;
  y[idx] = xb[0] * xb[(long)C * HW];
}

__global__ void k_avgpool(const float* __restrict__ x, float* __restrict__ a, long HW) {
  __shared__ float red[256];
  long base = (long)blockIdx.x * HW;
  float s = 0.f;
  for (long p = threadIdx.x; p < HW; p += 256) s += x[base + p];
  red[threadIdx.x] = s;
  __syncthreads();
  for (int st = 128; st > 0; st >>= 1) {
    if ((int)threadIdx.x < st) red[threadIdx.x] += red[threadIdx.x + st];
    __syncthreads();
  }
  if (threadIdx.x == 0) a[blockIdx.x] = red[0] / (float)HW;
}

__global__ void k_mulb(const float* __restrict__ x, const float* __restrict__ g,
                       float* __restrict__ y, long HW, long total) {
  long idx = (long)blockIdx.x * 256 + threadIdx.x;
  if (idx >= total) return;
  y[idx] = x[idx] * g[idx / HW];
}

__global__ void k_scale_add(const float* __restrict__ a, const float* __restrict__ x,
                            const float* __restrict__ sc, float* __restrict__ y,
                            int C, long HW, long total) {
  long idx = (long)blockIdx.x * 256 + threadIdx.x;
  if (idx >= total) return;
  y[idx] = a[idx] + x[idx] * sc[(idx / HW) % C];
}

__global__ void k_add(const float* __restrict__ x, const float* __restrict__ s,
                      float* __restrict__ y, long total) {
  long idx = (long)blockIdx.x * 256 + threadIdx.x;
  if (idx >= total) return;
  y[idx] = x[idx] + s[idx];
}

__global__ void k_pixel_shuffle(const float* __restrict__ x, float* __restrict__ y,
                                int C, int H, int W, long total) {
  long idx = (long)blockIdx.x * 256 + threadIdx.x;
  if (idx >= total) return;
  int Wo = 2 * W, Ho = 2 * H;
  int wo = idx % Wo;
  long r = idx / Wo;
  int ho = r % Ho; r /= Ho;
  int c = r % C;
  long b = r / C;
  y[idx] = x[(((long)b * 4 * C + c * 4 + (ho & 1) * 2 + (wo & 1)) * H + (ho >> 1)) * W +
             (wo >> 1)];
}

__global__ void k_down2x2(const float* __restrict__ x, const float* __restrict__ w,
                          const float* __restrict__ bias, float* __restrict__ y,
                          int I, int O, int Ho, int Wo, long total) {
  long idx = (long)blockIdx.x * 256 + threadIdx.x;
  if (idx >= total) return;
  int wo = idx % Wo;
  long r = idx / Wo;
  int ho = r % Ho; r /= Ho;
  int o = r % O;
  long b = r / O;
  int Wi = Wo * 2, Hi = Ho * 2;
  float s = bias[o];
  for (int i = 0; i < I; ++i) {
    const float* xp = x + (((long)b * I + i) * Hi + ho * 2) * Wi + wo * 2;
    const float* wp = w + ((long)o * I + i) * 4;
    s += xp[0] * wp[0] + xp[1] * wp[1] + xp[Wi] * wp[2] + xp[Wi + 1] * wp[3];
  }
  y[idx] = s;
}

__global__ void k_conv3x3_small(const float* __restrict__ x, const float* __restrict__ w,
                                const float* __restrict__ bias,
                                const float* __restrict__ res, float* __restrict__ y,
                                int I, int O, int H, int W, long total) {
  long idx = (long)blockIdx.x * 256 + threadIdx.x;
  if (idx >= total) return;
  int ww = idx % W;
  long r = idx / W;
  int hh = r % H; r /= H;
  int o = r % O;
  long b = r / O;
  float s = bias[o];
  for (int i = 0; i < I; ++i)
    for (int dy = 0; dy < 3; ++dy) {
      int h2 = hh + dy - 1;
      if (h2 < 0 || h2 >= H) continue;
      for (int dx = 0; dx < 3; ++dx) {
        int w2 = ww + dx - 1;
        if (w2 < 0 || w2 >= W) continue;
        s += x[(((long)b * I + i) * H + h2) * W + w2] *
             w[(((long)o * I + i) * 3 + dy) * 3 + dx];
      }
    }
  if (res) s += res[idx];
  y[idx] = s;
}

__global__ void k_f2h(const float* __restrict__ x, _Float16* __restrict__ y, long n) {
  long idx = (long)blockIdx.x * 256 + threadIdx.x;
  if (idx < n) y[idx] = (_Float16)x[idx];
}

// ============================ host orchestration ============================

#define EW(total) dim3((unsigned)(((total) + 255) / 256))

struct BlockP {
  const float *n1w, *n1b, *n2w, *n2b, *c1w, *c1b, *c2w, *c2b, *scaw, *scab,
              *c3w, *c3b, *c4w, *c4b, *c5w, *c5b, *beta, *gamma;
};

static BlockP mkblock(void* const* d, int base) {
  BlockP p;
  p.n1w = (const float*)d[base + 0];  p.n1b = (const float*)d[base + 1];
  p.n2w = (const float*)d[base + 2];  p.n2b = (const float*)d[base + 3];
  p.c1w = (const float*)d[base + 4];  p.c1b = (const float*)d[base + 5];
  p.c2w = (const float*)d[base + 6];  p.c2b = (const float*)d[base + 7];
  p.scaw = (const float*)d[base + 8]; p.scab = (const float*)d[base + 9];
  p.c3w = (const float*)d[base + 10]; p.c3b = (const float*)d[base + 11];
  p.c4w = (const float*)d[base + 12]; p.c4b = (const float*)d[base + 13];
  p.c5w = (const float*)d[base + 14]; p.c5b = (const float*)d[base + 15];
  p.beta = (const float*)d[base + 16]; p.gamma = (const float*)d[base + 17];
  return p;
}

static void run_expert1x1(hipStream_t s, const float* x, const float* v,
                          const float* Wexp, const float* bexp, _Float16* aggW,
                          float* aggB, float* y, int I, int O, int H, int W, int B) {
  int hp = H / PATCH, wp = W / PATCH, NP = B * hp * wp;
  dim3 ga(16, NP);
  k_agg_w<<<ga, 256, 0, s>>>(v, Wexp, bexp, aggW, aggB, O, I, 1, hp, wp);
  dim3 gc(NP, (O + 63) / 64);
  k_conv1x1_wmma<<<gc, 256, 0, s>>>(x, aggW, aggB, y, I, O, H, W, 0);
}

static void run_expert3x3(hipStream_t s, const float* x, const float* v,
                          const float* Wexp, const float* bexp, _Float16* aggW,
                          float* aggB, float* y, int I, int O, int H, int W, int B) {
  int hp = H / PATCH, wp = W / PATCH, NP = B * hp * wp;
  dim3 ga(64, NP);
  k_agg_w<<<ga, 256, 0, s>>>(v, Wexp, bexp, aggW, aggB, O, I, 9, hp, wp);
  dim3 gc(NP, (O + 63) / 64);
  k_conv3x3_wmma<<<gc, 256, 0, s>>>(x, aggW, aggB, y, I, O, H, W);
}

static void run_cmoe(hipStream_t s, float* x, const float* v, const BlockP& P,
                     int c, int H, int W, int B, float* tA, float* tB, float* tC,
                     _Float16* aggW, float* aggB, float* pool) {
  long HW = (long)H * W;
  long totC = (long)B * c * HW;
  dim3 gLN((unsigned)((HW + 255) / 256), B);
  // y-branch
  k_layernorm<<<gLN, 256, 0, s>>>(x, P.n1w, P.n1b, tA, c, HW);
  run_expert1x1(s, tA, v, P.c1w, P.c1b, aggW, aggB, tB, c, 2 * c, H, W, B);
  run_expert3x3(s, tB, v, P.c2w, P.c2b, aggW, aggB, tA, 2 * c, 2 * c, H, W, B);
  k_gate<<<EW(totC), 256, 0, s>>>(tA, tB, c, HW, totC);
  k_avgpool<<<dim3((unsigned)(B * c)), 256, 0, s>>>(tB, pool, HW);
  run_expert1x1(s, tB, v, P.scaw, P.scab, aggW, aggB, tC, c, c, H, W, B);
  k_mulb<<<EW(totC), 256, 0, s>>>(tC, pool, tC, HW, totC);
  run_expert1x1(s, tC, v, P.c3w, P.c3b, aggW, aggB, tA, c, c, H, W, B);
  k_scale_add<<<EW(totC), 256, 0, s>>>(x, tA, P.beta, x, c, HW, totC);
  // second half
  k_layernorm<<<gLN, 256, 0, s>>>(x, P.n2w, P.n2b, tA, c, HW);
  run_expert1x1(s, tA, v, P.c4w, P.c4b, aggW, aggB, tB, c, 2 * c, H, W, B);
  k_gate<<<EW(totC), 256, 0, s>>>(tB, tC, c, HW, totC);
  run_expert1x1(s, tC, v, P.c5w, P.c5b, aggW, aggB, tA, c, c, H, W, B);
  k_scale_add<<<EW(totC), 256, 0, s>>>(x, tA, P.gamma, x, c, HW, totC);
}

extern "C" void kernel_launch(void* const* d_in, const int* in_sizes, int n_in,
                              void* d_out, int out_size, void* d_ws, size_t ws_size,
                              hipStream_t stream) {
  (void)in_sizes; (void)n_in; (void)out_size; (void)ws_size;
  const int B = 4;
  const float* inp = (const float*)d_in[0];
  const float* v0 = (const float*)d_in[1];
  const float* v1 = (const float*)d_in[2];
  const float* v2 = (const float*)d_in[3];
  const float* intro_w = (const float*)d_in[4];
  const float* intro_b = (const float*)d_in[5];
  const float* end_w = (const float*)d_in[6];
  const float* end_b = (const float*)d_in[7];
  const float* down0_w = (const float*)d_in[8];
  const float* down0_b = (const float*)d_in[9];
  const float* down1_w = (const float*)d_in[10];
  const float* down1_b = (const float*)d_in[11];
  const float* up0_w = (const float*)d_in[12];
  const float* up1_w = (const float*)d_in[13];
  BlockP enc0 = mkblock(d_in, 14);
  BlockP enc1 = mkblock(d_in, 32);
  BlockP midp = mkblock(d_in, 50);
  BlockP dec0 = mkblock(d_in, 68);
  BlockP dec1 = mkblock(d_in, 86);

  // ---- workspace arena ----
  char* ws = (char*)d_ws;
  size_t off = 0;
  auto alloc = [&](size_t bytes) -> void* {
    void* p = ws + off;
    off += (bytes + 255) & ~(size_t)255;
    return p;
  };
  const size_t FBIG = (size_t)B * 64 * 256 * 256 * sizeof(float);  // 2c @ full res
  float* bufA = (float*)alloc(FBIG);
  float* bufB = (float*)alloc(FBIG);
  float* bufC = (float*)alloc(FBIG);
  float* bufD = (float*)alloc(FBIG);
  float* e0 = (float*)alloc((size_t)B * 32 * 256 * 256 * sizeof(float));
  float* e1 = (float*)alloc((size_t)B * 64 * 128 * 128 * sizeof(float));
  _Float16* aggW = (_Float16*)alloc((size_t)256 * 64 * 576 * sizeof(_Float16));
  float* aggB = (float*)alloc((size_t)256 * 256 * sizeof(float));
  float* pool = (float*)alloc((size_t)B * 128 * sizeof(float));

  const long HW0 = 256 * 256, HW1 = 128 * 128;
  long tot;

  // intro 3x3 (3 -> 32) + enc0
  tot = (long)B * 32 * HW0;
  k_conv3x3_small<<<EW(tot), 256, 0, stream>>>(inp, intro_w, intro_b, nullptr, e0,
                                               3, 32, 256, 256, tot);
  run_cmoe(stream, e0, v0, enc0, 32, 256, 256, B, bufA, bufB, bufC, aggW, aggB, pool);

  // down0 (32 -> 64, /2) + enc1
  tot = (long)B * 64 * HW1;
  k_down2x2<<<EW(tot), 256, 0, stream>>>(e0, down0_w, down0_b, e1, 32, 64, 128, 128, tot);
  run_cmoe(stream, e1, v1, enc1, 64, 128, 128, B, bufA, bufB, bufC, aggW, aggB, pool);

  // down1 (64 -> 128, /2) + mid
  tot = (long)B * 128 * 64 * 64;
  k_down2x2<<<EW(tot), 256, 0, stream>>>(e1, down1_w, down1_b, bufD, 64, 128, 64, 64, tot);
  run_cmoe(stream, bufD, v2, midp, 128, 64, 64, B, bufA, bufB, bufC, aggW, aggB, pool);

  // up0: static 1x1 (128 -> 256) via WMMA, pixel shuffle, skip add, dec0
  k_f2h<<<EW(256 * 128), 256, 0, stream>>>(up0_w, aggW, 256 * 128);
  {
    dim3 gu((unsigned)(B * 2 * 2), 4);
    k_conv1x1_wmma<<<gu, 256, 0, stream>>>(bufD, aggW, nullptr, bufA, 128, 256,
                                           64, 64, 1);
  }
  tot = (long)B * 64 * HW1;
  k_pixel_shuffle<<<EW(tot), 256, 0, stream>>>(bufA, bufB, 64, 64, 64, tot);
  k_add<<<EW(tot), 256, 0, stream>>>(bufB, e1, bufB, tot);
  run_cmoe(stream, bufB, v1, dec0, 64, 128, 128, B, bufA, bufC, bufD, aggW, aggB, pool);

  // up1: static 1x1 (64 -> 128) via WMMA, pixel shuffle, skip add, dec1
  k_f2h<<<EW(128 * 64), 256, 0, stream>>>(up1_w, aggW, 128 * 64);
  {
    dim3 gu((unsigned)(B * 4 * 4), 2);
    k_conv1x1_wmma<<<gu, 256, 0, stream>>>(bufB, aggW, nullptr, bufA, 64, 128,
                                           128, 128, 1);
  }
  tot = (long)B * 32 * HW0;
  k_pixel_shuffle<<<EW(tot), 256, 0, stream>>>(bufA, bufC, 32, 128, 128, tot);
  k_add<<<EW(tot), 256, 0, stream>>>(bufC, e0, bufC, tot);
  run_cmoe(stream, bufC, v0, dec1, 32, 256, 256, B, bufA, bufB, bufD, aggW, aggB, pool);

  // end 3x3 (32 -> 3) + global residual -> d_out
  tot = (long)B * 3 * HW0;
  k_conv3x3_small<<<EW(tot), 256, 0, stream>>>(bufC, end_w, end_b, inp,
                                               (float*)d_out, 32, 3, 256, 256, tot);
}
